// ESN_25348896981273
// MI455X (gfx1250) — matrix operand order repaired and checked by reading, not compiled, not run on hardware
//
#include <hip/hip_runtime.h>
#include <hip/hip_bf16.h>
#include <math.h>

typedef __attribute__((ext_vector_type(16))) _Float16 v16h;
typedef __attribute__((ext_vector_type(8)))  float    v8f;

static constexpr int Bd = 256;   // batch
static constexpr int Td = 100;   // time steps
static constexpr int Id = 1024;  // input dim
static constexpr int Hd = 2048;  // hidden dim
static constexpr int Od = 10;    // output dim

// ---------------------------------------------------------------------------
// Prep kernels
// ---------------------------------------------------------------------------
__global__ void cvt_f32_to_f16(const float* __restrict__ in,
                               _Float16* __restrict__ out, int n) {
    int i = blockIdx.x * blockDim.x + threadIdx.x;
    if (i < n) out[i] = (_Float16)in[i];
}

// in: [K][N] f32 row-major  ->  out: [N][K] f16 (K contiguous)
__global__ void transpose_to_f16(const float* __restrict__ in,
                                 _Float16* __restrict__ out, int K, int N) {
    int idx = blockIdx.x * blockDim.x + threadIdx.x;
    if (idx >= K * N) return;
    int n = idx / K;
    int k = idx % K;
    out[idx] = (_Float16)in[(size_t)k * N + n];
}

// ---------------------------------------------------------------------------
// One recurrence step:
//   h_out[B][H] = tanh( h_prev @ W  +  x[:,t,:] @ Win )
// Single GEMM with K = Hd + Id using V_WMMA_F32_16X16X32_F16.
// Data movement: double-buffered LDS tiles filled with
// GLOBAL_LOAD_ASYNC_TO_LDS_B128 (ASYNCcnt), overlapping the WMMAs.
// ---------------------------------------------------------------------------
__global__ __launch_bounds__(128) void esn_step(
    const _Float16* __restrict__ hprev,  // [B][H] f16
    const _Float16* __restrict__ xf,     // [B][T][I] f16
    const _Float16* __restrict__ Wt,     // [H][H] f16, K-major (W^T)
    const _Float16* __restrict__ Wint,   // [H][I] f16, K-major (Win^T)
    _Float16* __restrict__ hout,         // [B][H] f16
    int t)
{
    constexpr int BM = 64, BN = 64, BK = 32;
    constexpr int NIT = (Hd + Id) / BK;              // 96 K-iterations
    __shared__ __align__(16) _Float16 As[2][BM][BK]; // A rows (m), k contiguous
    __shared__ __align__(16) _Float16 Bs[2][BN][BK]; // B cols (n), k contiguous

    const int tid  = threadIdx.x;
    const int lane = tid & 31;
    const int wave = tid >> 5;          // 4 waves
    const int wr   = wave >> 1;         // wave row (0..1) -> 32-row strip
    const int wc   = wave & 1;          // wave col (0..1) -> 32-col strip
    const int g    = lane >> 4;         // lane group (0..1)
    const int lm   = lane & 15;         // row/col within 16

    const int m0 = blockIdx.y * BM;
    const int n0 = blockIdx.x * BN;

    // Async fill of one (A,B) tile pair into LDS buffer `buf`.
    // 64 rows x 32 f16 = 256 x 16B chunks per tile; 128 threads -> 2 apiece.
    auto issue_tile = [&](int kb, int buf) {
        for (int it = 0; it < 2; ++it) {
            int chunk = tid + it * 128;
            int r = chunk >> 2;            // row within tile
            int c = (chunk & 3) * 8;       // f16 offset within row
            const _Float16* sa;
            const _Float16* sb;
            if (kb < Hd) {
                sa = hprev + (size_t)(m0 + r) * Hd + kb + c;
                sb = Wt    + (size_t)(n0 + r) * Hd + kb + c;
            } else {
                sa = xf   + ((size_t)(m0 + r) * Td + t) * Id + (kb - Hd) + c;
                sb = Wint + (size_t)(n0 + r) * Id + (kb - Hd) + c;
            }
            unsigned la = (unsigned)(uintptr_t)&As[buf][r][c];
            unsigned lb = (unsigned)(uintptr_t)&Bs[buf][r][c];
            asm volatile("global_load_async_to_lds_b128 %0, %1, off"
                         :: "v"(la), "v"(sa) : "memory");
            asm volatile("global_load_async_to_lds_b128 %0, %1, off"
                         :: "v"(lb), "v"(sb) : "memory");
        }
    };

    v8f acc[2][2];
    for (int i = 0; i < 2; ++i)
        for (int j = 0; j < 2; ++j)
            acc[i][j] = (v8f){};

    union Frag { v16h v; uint4 u[2]; };

    // Prologue: fill buffer 0.
    issue_tile(0, 0);
    asm volatile("s_wait_asynccnt 0x0" ::: "memory");
    __syncthreads();

    for (int i = 0; i < NIT; ++i) {
        const int buf = i & 1;
        // Prefetch next tile into the other buffer while we compute.
        if (i + 1 < NIT) issue_tile((i + 1) * BK, buf ^ 1);

        // WMMA fragments per ISA VGPR layout (wave32):
        // A lane L: row = L%16, K in [8g,8g+8) then [16+8g,16+8g+8)
        // B lane L: col = L%16, K in [8g,8g+8) then [16+8g,16+8g+8)
        Frag a[2], b[2];
        for (int q = 0; q < 2; ++q) {
            int row = wr * 32 + q * 16 + lm;
            a[q].u[0] = *(const uint4*)&As[buf][row][8 * g];
            a[q].u[1] = *(const uint4*)&As[buf][row][16 + 8 * g];
            int col = wc * 32 + q * 16 + lm;
            b[q].u[0] = *(const uint4*)&Bs[buf][col][8 * g];
            b[q].u[1] = *(const uint4*)&Bs[buf][col][16 + 8 * g];
        }
        for (int q = 0; q < 2; ++q)
            for (int j = 0; j < 2; ++j)
                acc[q][j] = __builtin_amdgcn_wmma_f32_16x16x32_f16(
                    false, a[q].v, false, b[j].v,
                    (short)0, acc[q][j], false, false);

        // Drain the async fill of the next buffer, then rendezvous.
        asm volatile("s_wait_asynccnt 0x0" ::: "memory");
        __syncthreads();
    }

    // Epilogue: h = tanh(acc), store f16.
    // C/D layout: lane L, vgpr v -> (m_local = 8*g + v, n_local = L%16)
    for (int i = 0; i < 2; ++i)
        for (int j = 0; j < 2; ++j)
            for (int v = 0; v < 8; ++v) {
                int m = m0 + wr * 32 + i * 16 + g * 8 + v;
                int n = n0 + wc * 32 + j * 16 + lm;
                hout[(size_t)m * Hd + n] = (_Float16)tanhf(acc[i][j][v]);
            }
}

// ---------------------------------------------------------------------------
// Output projection: y[b][t][o] = sum_k Hall[t][b][k] * lin_w[o][k] + lin_b[o]
// (~1 GFLOP; lin_w is 80 KB -> lives in cache)
// ---------------------------------------------------------------------------
__global__ void esn_out(const _Float16* __restrict__ Hall,  // [T][B][H]
                        const float* __restrict__ lin_w,    // [O][H]
                        const float* __restrict__ lin_b,    // [O]
                        float* __restrict__ y)              // [B][T][O]
{
    int idx = blockIdx.x * blockDim.x + threadIdx.x;
    if (idx >= Bd * Td * Od) return;
    int o  = idx % Od;
    int bt = idx / Od;
    int t  = bt % Td;
    int b  = bt / Td;
    const _Float16* h = Hall + ((size_t)t * Bd + b) * Hd;
    const float*    w = lin_w + (size_t)o * Hd;
    float acc = lin_b[o];
    for (int k = 0; k < Hd; ++k) acc += (float)h[k] * w[k];
    y[idx] = acc;
}

// ---------------------------------------------------------------------------
extern "C" void kernel_launch(void* const* d_in, const int* in_sizes, int n_in,
                              void* d_out, int out_size, void* d_ws, size_t ws_size,
                              hipStream_t stream) {
    const float* x     = (const float*)d_in[0];  // [B][T][I]
    const float* h0    = (const float*)d_in[1];  // [B][H]
    const float* Win   = (const float*)d_in[2];  // [I][H]
    const float* W     = (const float*)d_in[3];  // [H][H]
    const float* lin_w = (const float*)d_in[4];  // [O][H]
    const float* lin_b = (const float*)d_in[5];  // [O]
    float* y = (float*)d_out;                    // [B][T][O]

    // Workspace carve-up (all sizes multiples of 256 bytes; ~171 MB total)
    char* ws = (char*)d_ws;
    size_t off = 0;
    _Float16* xf16 = (_Float16*)(ws + off); off += (size_t)Bd * Td * Id * 2; // 52.4 MB
    _Float16* Wt   = (_Float16*)(ws + off); off += (size_t)Hd * Hd * 2;      //  8.4 MB
    _Float16* Wint = (_Float16*)(ws + off); off += (size_t)Hd * Id * 2;      //  4.2 MB
    _Float16* h0f  = (_Float16*)(ws + off); off += (size_t)Bd * Hd * 2;      //  1.0 MB
    _Float16* Hall = (_Float16*)(ws + off); off += (size_t)Td * Bd * Hd * 2; // 104.9 MB

    // --- prep ---
    {
        int n = Bd * Td * Id;
        cvt_f32_to_f16<<<(n + 255) / 256, 256, 0, stream>>>(x, xf16, n);
    }
    {
        int n = Bd * Hd;
        cvt_f32_to_f16<<<(n + 255) / 256, 256, 0, stream>>>(h0, h0f, n);
    }
    {
        int n = Hd * Hd;  // W [H][H] -> Wt [n][k]
        transpose_to_f16<<<(n + 255) / 256, 256, 0, stream>>>(W, Wt, Hd, Hd);
    }
    {
        int n = Id * Hd;  // Win [I][H] -> Wint [n=H][k=I]
        transpose_to_f16<<<(n + 255) / 256, 256, 0, stream>>>(Win, Wint, Id, Hd);
    }

    // --- sequential recurrence: 100 dependent GEMM+tanh steps ---
    dim3 grid(Hd / 64, Bd / 64);  // 32 x 4 = 128 workgroups
    for (int t = 0; t < Td; ++t) {
        const _Float16* hp = (t == 0) ? h0f : (Hall + (size_t)(t - 1) * Bd * Hd);
        _Float16* ho = Hall + (size_t)t * Bd * Hd;
        esn_step<<<grid, 128, 0, stream>>>(hp, xf16, Wt, Wint, ho, t);
    }

    // --- output projection ---
    {
        int n = Bd * Td * Od;
        esn_out<<<(n + 255) / 256, 256, 0, stream>>>(Hall, lin_w, lin_b, y);
    }
}
